// RN_76175539962699
// MI455X (gfx1250) — matrix operand order repaired and checked by reading, not compiled, not run on hardware
//
#include <hip/hip_runtime.h>
#include <hip/hip_bf16.h>
#include <hip/hip_fp16.h>

typedef __attribute__((ext_vector_type(16))) _Float16 v16h;
typedef __attribute__((ext_vector_type(8)))  _Float16 v8h;
typedef __attribute__((ext_vector_type(8)))  float    v8f;

#define B_    4
#define S_    5
#define Q_    15
#define C_    256
#define D_    5
#define DD    25          // d*d pixels
#define CC    258         // channels + 2 coords
#define KD    128         // hidden1 width (WMMA K total)
#define N2    64          // hidden2 width (WMMA N total)
#define NPAIR (DD*DD)     // 625 pixel pairs
#define MT    ((NPAIR + 15)/16)   // 40 M-tiles of 16
#define MPW   (MT / 4)            // 10 M-tiles per wave

// ---------------------------------------------------------------------------
// Kernel 1a: Ainp[b,s,i,k] = inp_feat(b,s,i,:) . W1[0:258, k]   (f16 out)
// ---------------------------------------------------------------------------
__global__ void proj_input_kernel(const float* __restrict__ input_f,
                                  const float* __restrict__ W1,
                                  _Float16* __restrict__ Ainp) {
  __shared__ float feat[CC];
  const int row = blockIdx.x;        // (b*S+s)*DD + i
  const int i   = row % DD;
  const int bs  = row / DD;          // b*S+s
  const int tid = threadIdx.x;
  for (int c = tid; c < CC; c += blockDim.x) {
    float v;
    if (c < C_)       v = input_f[(bs * C_ + c) * DD + i];
    else if (c == C_) v = (float)(i / D_) * (1.0f / (float)D_);
    else              v = (float)(i % D_) * (1.0f / (float)D_);
    feat[c] = v;
  }
  __syncthreads();
  const int k = tid;
  float acc = 0.f;
  for (int c = 0; c < CC; ++c)
    acc = fmaf(feat[c], W1[c * KD + k], acc);
  Ainp[row * KD + k] = (_Float16)acc;
}

// ---------------------------------------------------------------------------
// Kernel 1b: Bq[b,q,j,k] = qry_feat(b,q,j,:) . W1[258:516, k] + b1[k]  (f16)
// ---------------------------------------------------------------------------
__global__ void proj_query_kernel(const float* __restrict__ query_f,
                                  const float* __restrict__ W1,
                                  const float* __restrict__ b1,
                                  _Float16* __restrict__ Bq) {
  __shared__ float feat[CC];
  const int row = blockIdx.x;        // (b*Q+q)*DD + j
  const int j   = row % DD;
  const int bq  = row / DD;          // b*Q+q
  const int tid = threadIdx.x;
  for (int c = tid; c < CC; c += blockDim.x) {
    float v;
    if (c < C_)       v = query_f[(bq * C_ + c) * DD + j];
    else if (c == C_) v = (float)(j / D_) * (1.0f / (float)D_);
    else              v = (float)(j % D_) * (1.0f / (float)D_);
    feat[c] = v;
  }
  __syncthreads();
  const int k = tid;
  float acc = b1[k];
  for (int c = 0; c < CC; ++c)
    acc = fmaf(feat[c], W1[(CC + c) * KD + k], acc);
  Bq[row * KD + k] = (_Float16)acc;
}

// ---------------------------------------------------------------------------
// Kernel 2: WMMA core. One workgroup per (b,q,s); 4 wave32s.
// Each wave owns ALL 64 output columns (4 B-fragment sets kept in VGPRs, 4
// independent WMMA accumulator chains sharing one A fragment) and 1/4 of the
// M-tiles. launch_bounds(128,1) gives the allocator the full VGPR file so the
// 16 B-fragments never spill. Tail tile 39 is handled in-loop: A rows >= 625
// route to a zeroed pad row; accumulation is masked only on the last tile
// (wave-uniform scalar branch, no code duplication).
// ---------------------------------------------------------------------------
__global__ void __launch_bounds__(128, 1)
relnet_wmma_kernel(const _Float16* __restrict__ Ainp,
                   const _Float16* __restrict__ Bq,
                   const float* __restrict__ W2,
                   const float* __restrict__ b2,
                   float* __restrict__ xg) {
  __shared__ _Float16 Al[(DD + 1) * KD];   // 6.5 KB (row DD zeroed)
  __shared__ _Float16 Bl[(DD + 1) * KD];   // 6.5 KB
  __shared__ float part[512];              // [wave][ntile][lane]

  const int blk = blockIdx.x;       // ((b*Q)+q)*S + s
  const int s   = blk % S_;
  const int bq  = blk / S_;         // b*Q+q
  const int b   = bq / Q_;
  const int tid = threadIdx.x;

  // Stage f16 rows as dwords; zero the pad row.
  const unsigned* __restrict__ As32 =
      (const unsigned*)(Ainp + (size_t)(b * S_ + s) * DD * KD);
  const unsigned* __restrict__ Bs32 =
      (const unsigned*)(Bq + (size_t)bq * DD * KD);
  unsigned* Al32 = (unsigned*)Al;
  unsigned* Bl32 = (unsigned*)Bl;
  for (int idx = tid; idx < (DD + 1) * KD / 2; idx += blockDim.x) {
    const bool inr = idx < DD * KD / 2;
    Al32[idx] = inr ? As32[idx] : 0u;
    Bl32[idx] = inr ? Bs32[idx] : 0u;
  }
  __syncthreads();

  const int wave  = tid >> 5;
  const int lane  = tid & 31;
  const int lmod  = lane & 15;
  const int lhalf = lane >> 4;

  float bias[4];
  #pragma unroll
  for (int nt = 0; nt < 4; ++nt) bias[nt] = b2[nt * 16 + lmod];

  // B fragments (W2, f32 -> f16), ISA 16-bit B layout (32x16):
  // lane half h holds K=kbase+{8h+0..7, 16+8h+0..7}; N=lane%16.
  v16h bf[4][4];                     // [ntile][kstep]
  #pragma unroll
  for (int nt = 0; nt < 4; ++nt) {
    const int ncol = nt * 16 + lmod;
    #pragma unroll
    for (int kk = 0; kk < 4; ++kk) {
      #pragma unroll
      for (int e = 0; e < 16; ++e) {
        const int kidx = kk * 32 + (e >> 3) * 16 + lhalf * 8 + (e & 7);
        bf[nt][kk][e] = (_Float16)W2[kidx * N2 + ncol];
      }
    }
  }

  // A-fragment builder. ISA 16-bit A layout (16x32): lane half h holds
  // K = {8h+0..7, 16+8h+0..7} per 32-K step -> two aligned 16B chunks.
  const v8h zero8 = {};
  auto make_afrag = [&](const _Float16* arow, const _Float16* brow,
                        int kk) -> v16h {
    const v8h a1 = *(const v8h*)(arow + kk * 32 + lhalf * 8);
    const v8h a2 = *(const v8h*)(arow + kk * 32 + 16 + lhalf * 8);
    const v8h r1 = *(const v8h*)(brow + kk * 32 + lhalf * 8);
    const v8h r2 = *(const v8h*)(brow + kk * 32 + 16 + lhalf * 8);
    const v8h lo = __builtin_elementwise_max(a1 + r1, zero8);   // packed relu
    const v8h hi = __builtin_elementwise_max(a2 + r2, zero8);
    return __builtin_shufflevector(lo, hi, 0, 1, 2, 3, 4, 5, 6, 7,
                                   8, 9, 10, 11, 12, 13, 14, 15);
  };

  float acc0 = 0.f, acc1 = 0.f, acc2 = 0.f, acc3 = 0.f;
  const int mstart = wave * MPW;
  const int mend   = mstart + MPW;

  for (int mt = mstart; mt < mend; ++mt) {
    const int p  = mt * 16 + lmod;
    const int pi = (p < NPAIR) ? (p / DD) : DD;   // DD -> zero pad row
    const int pj = (p < NPAIR) ? (p % DD) : DD;
    const _Float16* __restrict__ arow = &Al[pi * KD];
    const _Float16* __restrict__ brow = &Bl[pj * KD];

    v8f c0 = {}, c1 = {}, c2 = {}, c3 = {};
    #pragma unroll
    for (int kk = 0; kk < 4; ++kk) {
      const v16h a = make_afrag(arow, brow, kk);
      c0 = __builtin_amdgcn_wmma_f32_16x16x32_f16(false, a, false, bf[0][kk],
                                                  (short)0, c0, false, false);
      c1 = __builtin_amdgcn_wmma_f32_16x16x32_f16(false, a, false, bf[1][kk],
                                                  (short)0, c1, false, false);
      c2 = __builtin_amdgcn_wmma_f32_16x16x32_f16(false, a, false, bf[2][kk],
                                                  (short)0, c2, false, false);
      c3 = __builtin_amdgcn_wmma_f32_16x16x32_f16(false, a, false, bf[3][kk],
                                                  (short)0, c3, false, false);
    }

    if (mt != MT - 1) {                 // all 16 rows valid (wave-uniform)
      #pragma unroll
      for (int v = 0; v < 8; ++v) {
        float h0 = c0[v] + bias[0];
        float h1 = c1[v] + bias[1];
        float h2 = c2[v] + bias[2];
        float h3 = c3[v] + bias[3];
        acc0 += h0 > 0.f ? h0 : 0.f;
        acc1 += h1 > 0.f ? h1 : 0.f;
        acc2 += h2 > 0.f ? h2 : 0.f;
        acc3 += h3 > 0.f ? h3 : 0.f;
      }
    } else {                            // tail tile: mask rows >= 625
      #pragma unroll
      for (int v = 0; v < 8; ++v) {
        const int pr = mt * 16 + v + lhalf * 8;
        float h0 = c0[v] + bias[0];
        float h1 = c1[v] + bias[1];
        float h2 = c2[v] + bias[2];
        float h3 = c3[v] + bias[3];
        h0 = h0 > 0.f ? h0 : 0.f;
        h1 = h1 > 0.f ? h1 : 0.f;
        h2 = h2 > 0.f ? h2 : 0.f;
        h3 = h3 > 0.f ? h3 : 0.f;
        acc0 += (pr < NPAIR) ? h0 : 0.f;
        acc1 += (pr < NPAIR) ? h1 : 0.f;
        acc2 += (pr < NPAIR) ? h2 : 0.f;
        acc3 += (pr < NPAIR) ? h3 : 0.f;
      }
    }
  }

  part[(wave * 4 + 0) * 32 + lane] = acc0;
  part[(wave * 4 + 1) * 32 + lane] = acc1;
  part[(wave * 4 + 2) * 32 + lane] = acc2;
  part[(wave * 4 + 3) * 32 + lane] = acc3;
  __syncthreads();

  // Column n: sum over all 4 waves, both lane halves, N-tile n/16.
  if (tid < N2) {
    const int tileN = tid >> 4;
    const int lm    = tid & 15;
    float r = 0.f;
    #pragma unroll
    for (int w = 0; w < 4; ++w) {
      const float* pp = &part[(w * 4 + tileN) * 32];
      r += pp[lm] + pp[16 + lm];
    }
    xg[(size_t)blk * N2 + tid] = r;
  }
}

// ---------------------------------------------------------------------------
// Kernel 3: f MLP head + sigmoid + label compare + MSE loss reduction.
// out[0] = loss, out[1..300] = scores (flattened [B,Q,S,1]).
// ---------------------------------------------------------------------------
__global__ void head_kernel(const float* __restrict__ xg,
                            const float* __restrict__ Wf1, const float* __restrict__ bf1,
                            const float* __restrict__ Wf2, const float* __restrict__ bf2,
                            const int* __restrict__ input_y,
                            const int* __restrict__ query_y,
                            float* __restrict__ out) {
  __shared__ float red[512];
  const int t = threadIdx.x;
  float sq = 0.f;
  if (t < B_ * Q_ * S_) {
    const float* __restrict__ x = xg + (size_t)t * N2;
    float h[16];
    #pragma unroll
    for (int kk = 0; kk < 16; ++kk) {
      float a = bf1[kk];
      for (int c = 0; c < N2; ++c) a = fmaf(x[c], Wf1[c * 16 + kk], a);
      h[kk] = a > 0.f ? a : 0.f;
    }
    float z = bf2[0];
    #pragma unroll
    for (int kk = 0; kk < 16; ++kk) z = fmaf(h[kk], Wf2[kk], z);
    const float score = 1.f / (1.f + __expf(-z));
    const int s  = t % S_;
    const int bq = t / S_;          // b*Q+q
    const int b  = bq / Q_;
    const float label = (input_y[b * S_ + s] == query_y[bq]) ? 1.f : 0.f;
    const float df = label - score;
    sq = df * df;
    out[1 + t] = score;
  }
  red[t] = sq;
  __syncthreads();
  for (int off = 256; off > 0; off >>= 1) {
    if (t < off) red[t] += red[t + off];
    __syncthreads();
  }
  if (t == 0) out[0] = red[0];
}

// ---------------------------------------------------------------------------
extern "C" void kernel_launch(void* const* d_in, const int* in_sizes, int n_in,
                              void* d_out, int out_size, void* d_ws, size_t ws_size,
                              hipStream_t stream) {
  const float* input_f = (const float*)d_in[0];
  const float* query_f = (const float*)d_in[1];
  const int*   input_y = (const int*)d_in[2];
  const int*   query_y = (const int*)d_in[3];
  const float* W1      = (const float*)d_in[4];
  const float* b1      = (const float*)d_in[5];
  const float* W2      = (const float*)d_in[6];
  const float* b2      = (const float*)d_in[7];
  const float* Wf1     = (const float*)d_in[8];
  const float* bf1     = (const float*)d_in[9];
  const float* Wf2     = (const float*)d_in[10];
  const float* bf2     = (const float*)d_in[11];
  float* out = (float*)d_out;

  _Float16* Ainp = (_Float16*)d_ws;                          // 500*128 halves
  _Float16* Bq   = Ainp + (size_t)B_ * S_ * DD * KD;         // 1500*128 halves
  float*    xg   = (float*)(Bq + (size_t)B_ * Q_ * DD * KD); // 300*64 floats

  proj_input_kernel<<<B_ * S_ * DD, KD, 0, stream>>>(input_f, W1, Ainp);
  proj_query_kernel<<<B_ * Q_ * DD, KD, 0, stream>>>(query_f, W1, b1, Bq);
  relnet_wmma_kernel<<<B_ * Q_ * S_, 128, 0, stream>>>(Ainp, Bq, W2, b2, xg);
  head_kernel<<<1, 512, 0, stream>>>(xg, Wf1, bf1, Wf2, bf2,
                                     input_y, query_y, out);
}
